// Generator_2078764171989
// MI455X (gfx1250) — matrix operand order repaired and checked by reading, not compiled, not run on hardware
//
#include <hip/hip_runtime.h>
#include <math.h>

typedef __attribute__((ext_vector_type(16))) _Float16 v16h;
typedef __attribute__((ext_vector_type(8)))  _Float16 v8h;
typedef __attribute__((ext_vector_type(8)))  float    v8f;

#define LEAKY(v) ((v) > 0.f ? (v) : 0.2f * (v))

// ---------------------------------------------------------------------------
// Broadcast initial constant [1,512,4,4] NCHW f32 -> act [16,4,4,512] NHWC f16
// ---------------------------------------------------------------------------
__global__ __launch_bounds__(256) void k_init(const float* __restrict__ ic,
                                              _Float16* __restrict__ act) {
  int idx = blockIdx.x * 256 + threadIdx.x;
  if (idx >= 16 * 4 * 4 * 512) return;
  int c = idx & 511;
  int t = idx >> 9;
  int x = t & 3;  t >>= 2;
  int y = t & 3;
  act[idx] = (_Float16)ic[(c * 4 + y) * 4 + x];
}

// ---------------------------------------------------------------------------
// Style vector: s[b,c] = dot(wlat[b,:], tsw[c,:])/sqrt(512) + tsb[c]
// Stage latent row in LDS; emit f32 (for demod / to_rgb) and f16*(mod scale)
// ---------------------------------------------------------------------------
__global__ __launch_bounds__(256) void k_style(const float* __restrict__ wlat,
                                               const float* __restrict__ tsw,
                                               const float* __restrict__ tsb,
                                               float* __restrict__ s32,
                                               _Float16* __restrict__ s16,
                                               int Cin, float mod_scale) {
  __shared__ float lw[512];
  int b = blockIdx.x;
  for (int d = threadIdx.x; d < 512; d += 256) lw[d] = wlat[b * 512 + d];
  __syncthreads();
  const float cl = 0.044194173824159216f;  // 1/sqrt(512)
  for (int ci = threadIdx.x; ci < Cin; ci += 256) {
    const float* wr = tsw + (size_t)ci * 512;
    float acc = 0.f;
    for (int d = 0; d < 512; ++d) acc += lw[d] * wr[d];
    float sv = acc * cl + tsb[ci];
    s32[b * Cin + ci] = sv;
    s16[b * Cin + ci] = (_Float16)(sv * mod_scale);
  }
}

// ---------------------------------------------------------------------------
// demod[b,co] = rsqrt(mscale^2 * sum_ci s^2 * sum_k w^2 + eps)
// ---------------------------------------------------------------------------
__global__ __launch_bounds__(256) void k_demod(const float* __restrict__ w,
                                               const float* __restrict__ s32,
                                               float* __restrict__ dmod,
                                               int Cin, int Cout, float mscale) {
  int idx = blockIdx.x * 256 + threadIdx.x;
  if (idx >= 16 * Cout) return;
  int b = idx / Cout, co = idx % Cout;
  const float* sp = s32 + b * Cin;
  float acc = 0.f;
  for (int ci = 0; ci < Cin; ++ci) {
    const float* wp = w + ((size_t)co * Cin + ci) * 9;
    float wsum = 0.f;
    for (int k = 0; k < 9; ++k) { float wv = wp[k]; wsum += wv * wv; }
    float sv = sp[ci];
    acc += sv * sv * wsum;
  }
  dmod[idx] = rsqrtf(mscale * mscale * acc + 1e-8f);
}

// ---------------------------------------------------------------------------
// Repack conv weight f32 [co][ci][3][3] -> f16 [co][kpos][ci] (K kpos-major)
// ---------------------------------------------------------------------------
__global__ __launch_bounds__(256) void k_pack(const float* __restrict__ w,
                                              _Float16* __restrict__ wp,
                                              int Cin, long total) {
  long idx = (long)blockIdx.x * 256 + threadIdx.x;
  if (idx >= total) return;
  long kpos = idx % 9;
  long t = idx / 9;
  long ci = t % Cin;
  long co = t / Cin;
  wp[(co * 9 + kpos) * Cin + ci] = (_Float16)w[idx];
}

// ---------------------------------------------------------------------------
// Modulated 3x3 conv as implicit GEMM on v_wmma_f32_16x16x32_f16.
// One wave32 computes a 16(cout) x (16*NT)(pixels) tile. K = 9*Cin, chunked by
// 32 contiguous channels (NHWC acts, kpos-major packed weights). A fragment is
// reused across NT pixel tiles. Halo handling is branch-free: OOB addresses
// are clamped and the contribution is zeroed by a multiplicative f16 mask
// folded into the modulation pk_mul, keeping EXEC uniform for WMMA.
// Epilogue: *demod + scale_noise*noise + bias, leaky 0.2, store f16 NHWC.
// ---------------------------------------------------------------------------
template <int NT>
__global__ __launch_bounds__(32) void k_conv3x3(
    const _Float16* __restrict__ xin, const _Float16* __restrict__ s16,
    const _Float16* __restrict__ wpk, const float* __restrict__ dmod,
    const float* __restrict__ noise, const float* __restrict__ snoise,
    const float* __restrict__ bias, _Float16* __restrict__ out,
    int H, int W, int Cin, int Cout) {
  const int b = blockIdx.z;
  const int cotile = blockIdx.y;
  const int lane = threadIdx.x;
  const int m = lane & 15;      // A row (cout) / B column (pixel)
  const int g = lane >> 4;      // half-wave group selects K sub-fragment

  int py[NT], px[NT];
#pragma unroll
  for (int t = 0; t < NT; ++t) {
    const int p = blockIdx.x * (16 * NT) + 16 * t + m;
    py[t] = p / W;
    px[t] = p % W;
  }

  const _Float16* sb = s16 + (size_t)b * Cin;
  const _Float16* wbase = wpk + (size_t)(cotile * 16 + m) * 9 * Cin;
  const _Float16* xb = xin + (size_t)b * H * W * Cin;

  v8f acc[NT];
#pragma unroll
  for (int t = 0; t < NT; ++t) acc[t] = (v8f){};

  for (int cc = 0; cc < Cin; cc += 32) {
    // modulation fragment: depends only on channel chunk -> hoisted
    const v8h sm0 = *(const v8h*)(sb + cc + g * 16);
    const v8h sm1 = *(const v8h*)(sb + cc + g * 16 + 8);
    __builtin_prefetch(wbase + cc + 32, 0, 1);  // global_prefetch_b8
#pragma unroll
    for (int kpos = 0; kpos < 9; ++kpos) {
      // A fragment: lane holds K = g*16 .. g*16+15 (contiguous channels)
      const _Float16* ap = wbase + kpos * Cin + cc;
      const v8h a0 = *(const v8h*)(ap + g * 16);
      const v8h a1 = *(const v8h*)(ap + g * 16 + 8);
      const v16h av = __builtin_shufflevector(a0, a1, 0, 1, 2, 3, 4, 5, 6, 7,
                                              8, 9, 10, 11, 12, 13, 14, 15);
      const int dy = kpos / 3 - 1, dx = kpos % 3 - 1;
#pragma unroll
      for (int t = 0; t < NT; ++t) {
        const int yy = py[t] + dy;
        const int xx = px[t] + dx;
        const bool inb = (yy >= 0) & (yy < H) & (xx >= 0) & (xx < W);
        const _Float16 mk = inb ? (_Float16)1.0f : (_Float16)0.0f;
        // B fragment: lane holds K = g*16 .. g*16+15 (contiguous channels)
        const _Float16* bp =
            xb + ((size_t)(inb ? yy : 0) * W + (inb ? xx : 0)) * Cin + cc +
            g * 16;
        v8h b0 = *(const v8h*)(bp);
        v8h b1 = *(const v8h*)(bp + 8);
        b0 = b0 * sm0 * mk;  // v_pk_mul_f16: modulation + halo mask
        b1 = b1 * sm1 * mk;
        const v16h bv = __builtin_shufflevector(b0, b1, 0, 1, 2, 3, 4, 5, 6, 7,
                                                8, 9, 10, 11, 12, 13, 14, 15);
        acc[t] = __builtin_amdgcn_wmma_f32_16x16x32_f16(
            false, av, false, bv, (short)0, acc[t], false, false);
      }
    }
  }

  const float sn = snoise[0];
#pragma unroll
  for (int t = 0; t < NT; ++t) {
    const float nz = sn * noise[((size_t)b * H + py[t]) * W + px[t]];
    _Float16* op = out + (((size_t)b * H + py[t]) * W + px[t]) * Cout;
#pragma unroll
    for (int r = 0; r < 8; ++r) {
      const int co = cotile * 16 + g * 8 + r;  // C layout: M = r + 8*g
      float v = acc[t][r] * dmod[b * Cout + co] + nz + bias[co];
      op[co] = (_Float16)LEAKY(v);
    }
  }
}

// ---------------------------------------------------------------------------
// Fused bilinear 2x upsample (half-pixel) + edge-padded 3x3 smooth (1,2,1)/4 x2
// ---------------------------------------------------------------------------
template <typename T>
__global__ __launch_bounds__(256) void k_upsample(const T* __restrict__ in,
                                                  T* __restrict__ out,
                                                  int H, int W, int C,
                                                  long total) {
  long idx = (long)blockIdx.x * 256 + threadIdx.x;
  if (idx >= total) return;
  const int c = idx % C;
  long t = idx / C;
  const int X = t % (2 * W);  t /= (2 * W);
  const int Y = t % (2 * H);
  const int b = t / (2 * H);
  const float kw[3] = {0.25f, 0.5f, 0.25f};
  float acc = 0.f;
  for (int j = 0; j < 3; ++j) {
    int Yq = Y - 1 + j;
    Yq = Yq < 0 ? 0 : (Yq > 2 * H - 1 ? 2 * H - 1 : Yq);
    float fy = Yq * 0.5f - 0.25f;
    int y0 = (int)floorf(fy);
    float wy = fy - (float)y0;
    int y1 = y0 + 1;
    y0 = y0 < 0 ? 0 : (y0 > H - 1 ? H - 1 : y0);
    y1 = y1 < 0 ? 0 : (y1 > H - 1 ? H - 1 : y1);
    for (int i = 0; i < 3; ++i) {
      int Xq = X - 1 + i;
      Xq = Xq < 0 ? 0 : (Xq > 2 * W - 1 ? 2 * W - 1 : Xq);
      float fx = Xq * 0.5f - 0.25f;
      int x0 = (int)floorf(fx);
      float wx = fx - (float)x0;
      int x1 = x0 + 1;
      x0 = x0 < 0 ? 0 : (x0 > W - 1 ? W - 1 : x0);
      x1 = x1 < 0 ? 0 : (x1 > W - 1 ? W - 1 : x1);
      const size_t rb = ((size_t)b * H + y0) * W;
      const size_t rt = ((size_t)b * H + y1) * W;
      float v00 = (float)in[(rb + x0) * C + c];
      float v01 = (float)in[(rb + x1) * C + c];
      float v10 = (float)in[(rt + x0) * C + c];
      float v11 = (float)in[(rt + x1) * C + c];
      float bl = (1.f - wy) * ((1.f - wx) * v00 + wx * v01) +
                 wy * ((1.f - wx) * v10 + wx * v11);
      acc += kw[j] * kw[i] * bl;
    }
  }
  out[idx] = (T)acc;
}

// ---------------------------------------------------------------------------
// to_rgb: 1x1 modulated conv (no demod), leaky, optional add of upsampled rgb
// ---------------------------------------------------------------------------
__global__ __launch_bounds__(256) void k_torgb(
    const _Float16* __restrict__ act, const float* __restrict__ s32,
    const float* __restrict__ wrgb, const float* __restrict__ bias,
    const float* __restrict__ base, float* __restrict__ out,
    int HW, int Cin, float scale, long total) {
  long idx = (long)blockIdx.x * 256 + threadIdx.x;
  if (idx >= total) return;
  const int b = idx / HW;
  const _Float16* ap = act + idx * (long)Cin;
  const float* sp = s32 + b * Cin;
  float a0 = 0.f, a1 = 0.f, a2 = 0.f;
  for (int c = 0; c < Cin; ++c) {
    float xv = (float)ap[c] * sp[c];
    a0 += xv * wrgb[c];
    a1 += xv * wrgb[Cin + c];
    a2 += xv * wrgb[2 * Cin + c];
  }
  float r0 = LEAKY(a0 * scale + bias[0]);
  float r1 = LEAKY(a1 * scale + bias[1]);
  float r2 = LEAKY(a2 * scale + bias[2]);
  if (base) {
    r0 += base[idx * 3 + 0];
    r1 += base[idx * 3 + 1];
    r2 += base[idx * 3 + 2];
  }
  out[idx * 3 + 0] = r0;
  out[idx * 3 + 1] = r1;
  out[idx * 3 + 2] = r2;
}

// ---------------------------------------------------------------------------
// NHWC f32 rgb -> NCHW f32 output
// ---------------------------------------------------------------------------
__global__ __launch_bounds__(256) void k_final(const float* __restrict__ rgb,
                                               float* __restrict__ out) {
  long idx = (long)blockIdx.x * 256 + threadIdx.x;
  if (idx >= 16L * 3 * 256 * 256) return;
  const int x = idx & 255;
  long t = idx >> 8;
  const int y = t & 255;  t >>= 8;
  const int j = t % 3;
  const int b = t / 3;
  out[idx] = rgb[(((long)b * 256 + y) * 256 + x) * 3 + j];
}

// ===========================================================================
extern "C" void kernel_launch(void* const* d_in, const int* in_sizes, int n_in,
                              void* d_out, int out_size, void* d_ws,
                              size_t ws_size, hipStream_t stream) {
  (void)in_sizes; (void)n_in; (void)out_size; (void)ws_size;
  const int F[7] = {512, 512, 512, 256, 128, 64, 32};

  // ---- inputs (jax pytree order: sorted dict keys, lists in order) --------
  const float* wlat = (const float*)d_in[0];             // [7,16,512]
  // noises: d_in[1]=n0, d_in[2i]=n{i}a, d_in[2i+1]=n{i}b
  // params: blocks[0..5] x {sb1(5), sb2(5), to_rgb(4)} then ic, sb0, trgb0
  const int PBASE = 14;
  const int IC = PBASE + 84;                             // initial_constant
  auto Pf = [&](int i) { return (const float*)d_in[i]; };

  // ---- workspace layout ---------------------------------------------------
  char* wsb = (char*)d_ws;
  const size_t ACT_BYTES = 16ull * 256 * 256 * 64 * 2;   // 134 MB (f16)
  const size_t RGB_BYTES = 16ull * 256 * 256 * 3 * 4;    // 12.6 MB (f32)
  const size_t WPK_BYTES = 512ull * 9 * 512 * 2;         // 4.7 MB (f16)
  _Float16* actA = (_Float16*)wsb;
  _Float16* actB = (_Float16*)(wsb + ACT_BYTES);
  float* rgbA = (float*)(wsb + 2 * ACT_BYTES);
  float* rgbB = (float*)(wsb + 2 * ACT_BYTES + RGB_BYTES);
  _Float16* wpk = (_Float16*)(wsb + 2 * ACT_BYTES + 2 * RGB_BYTES);
  float* s32 = (float*)(wsb + 2 * ACT_BYTES + 2 * RGB_BYTES + WPK_BYTES);
  _Float16* s16 = (_Float16*)((char*)s32 + 16 * 512 * 4);
  float* dmod = (float*)((char*)s16 + 16 * 512 * 2);

  _Float16* cur = actA;
  _Float16* nxt = actB;

  auto style_conv = [&](int tsw, int tsb, int convw, int sn, int bias,
                        const float* wrow, const float* noise,
                        const _Float16* xin, _Float16* xout, int H, int Cin,
                        int Cout) {
    const float ms = 1.0f / sqrtf((float)(Cin * 9));
    k_style<<<16, 256, 0, stream>>>(wrow, Pf(tsw), Pf(tsb), s32, s16, Cin, ms);
    int dt = 16 * Cout;
    k_demod<<<(dt + 255) / 256, 256, 0, stream>>>(Pf(convw), s32, dmod, Cin,
                                                  Cout, ms);
    long wn = (long)Cout * Cin * 9;
    k_pack<<<(int)((wn + 255) / 256), 256, 0, stream>>>(Pf(convw), wpk, Cin, wn);
    const int HW = H * H;
    if (HW >= 32) {
      dim3 g(HW / 32, Cout / 16, 16);
      k_conv3x3<2><<<g, 32, 0, stream>>>(xin, s16, wpk, dmod, noise, Pf(sn),
                                         Pf(bias), xout, H, H, Cin, Cout);
    } else {
      dim3 g(HW / 16, Cout / 16, 16);
      k_conv3x3<1><<<g, 32, 0, stream>>>(xin, s16, wpk, dmod, noise, Pf(sn),
                                         Pf(bias), xout, H, H, Cin, Cout);
    }
  };
  auto to_rgb = [&](int tsw, int tsb, int convw, int bias, const float* wrow,
                    const _Float16* xin, const float* base, float* rout, int H,
                    int Cin) {
    k_style<<<16, 256, 0, stream>>>(wrow, Pf(tsw), Pf(tsb), s32, s16, Cin, 1.f);
    long tot = 16L * H * H;
    k_torgb<<<(int)((tot + 255) / 256), 256, 0, stream>>>(
        xin, s32, Pf(convw), Pf(bias), base, rout, H * H, Cin,
        1.0f / sqrtf((float)Cin), tot);
  };

  // ---- initial block at 4x4 ----------------------------------------------
  k_init<<<(16 * 4 * 4 * 512 + 255) / 256, 256, 0, stream>>>(Pf(IC), cur);
  // sb0: {bias=IC+1, conv=IC+2, sn=IC+3, tsb=IC+4, tsw=IC+5}
  style_conv(IC + 5, IC + 4, IC + 2, IC + 3, IC + 1, wlat,
             (const float*)d_in[1], cur, nxt, 4, 512, 512);
  { _Float16* t = cur; cur = nxt; nxt = t; }
  // trgb0: {bias=IC+6, conv=IC+7, tsb=IC+8, tsw=IC+9}
  to_rgb(IC + 9, IC + 8, IC + 7, IC + 6, wlat, cur, nullptr, rgbA, 4, 512);

  // ---- blocks i = 1..6 ----------------------------------------------------
  for (int i = 1; i <= 6; ++i) {
    const int Hp = 4 << (i - 1), H = 4 << i;
    const int Cp = F[i - 1], C = F[i];
    const float* wrow = wlat + (size_t)i * 16 * 512;
    const int bb = PBASE + 14 * (i - 1);
    // x = upsample(x)
    long tot = 16L * H * H * Cp;
    k_upsample<_Float16><<<(int)((tot + 255) / 256), 256, 0, stream>>>(
        cur, nxt, Hp, Hp, Cp, tot);
    { _Float16* t = cur; cur = nxt; nxt = t; }
    // sb1: leaves {bias+0, conv+1, sn+2, tsb+3, tsw+4}
    style_conv(bb + 4, bb + 3, bb + 1, bb + 2, bb + 0, wrow,
               (const float*)d_in[2 * i], cur, nxt, H, Cp, C);
    { _Float16* t = cur; cur = nxt; nxt = t; }
    // sb2: leaves {bias+5, conv+6, sn+7, tsb+8, tsw+9}
    style_conv(bb + 9, bb + 8, bb + 6, bb + 7, bb + 5, wrow,
               (const float*)d_in[2 * i + 1], cur, nxt, H, C, C);
    { _Float16* t = cur; cur = nxt; nxt = t; }
    // rgb = upsample(rgb) + to_rgb(x): leaves {bias+10, conv+11, tsb+12, tsw+13}
    long rt = 16L * H * H * 3;
    k_upsample<float><<<(int)((rt + 255) / 256), 256, 0, stream>>>(
        rgbA, rgbB, Hp, Hp, 3, rt);
    to_rgb(bb + 13, bb + 12, bb + 11, bb + 10, wrow, cur, rgbB, rgbA, H, C);
  }

  k_final<<<(int)((16L * 3 * 256 * 256 + 255) / 256), 256, 0, stream>>>(
      rgbA, (float*)d_out);
}
